// FlowModule_48163763257801
// MI455X (gfx1250) — matrix operand order repaired and checked by reading, not compiled, not run on hardware
//
#include <hip/hip_runtime.h>
#include <hip/hip_bf16.h>

typedef __attribute__((ext_vector_type(16))) __bf16 v16bf;
typedef __attribute__((ext_vector_type(8)))  float  v8f;

static constexpr int   kNB  = 4;
static constexpr int   kN   = 4096;
static constexpr int   kK   = 32;
static constexpr float kCnt = (float)(4 * 4096 * 32);

// ---------------------------------------------------------------------------
// KNN: one query point per thread, candidates staged via LDS, top-32 kept in
// registers (fully-unrolled replace + rescan so arrays stay in VGPRs).
// ---------------------------------------------------------------------------
__global__ __launch_bounds__(256) void knn_kernel(const float* __restrict__ pos1,
                                                  const float* __restrict__ pos2,
                                                  int* __restrict__ outidx) {
  __shared__ float sx[256], sy[256], sz[256];
  const int b = blockIdx.x >> 4;                       // 16 blocks per batch
  const int n = ((blockIdx.x & 15) << 8) + threadIdx.x;
  const float qx = pos1[(b * 3 + 0) * kN + n];
  const float qy = pos1[(b * 3 + 1) * kN + n];
  const float qz = pos1[(b * 3 + 2) * kN + n];

  float bd[kK];
  int   bi[kK];
#pragma unroll
  for (int j = 0; j < kK; ++j) { bd[j] = 3.0e38f; bi[j] = 0; }
  float wd = 3.0e38f;
  int wpos = 0;

  for (int m0 = 0; m0 < kN; m0 += 256) {
    __syncthreads();
    sx[threadIdx.x] = pos2[(b * 3 + 0) * kN + m0 + threadIdx.x];
    sy[threadIdx.x] = pos2[(b * 3 + 1) * kN + m0 + threadIdx.x];
    sz[threadIdx.x] = pos2[(b * 3 + 2) * kN + m0 + threadIdx.x];
    __syncthreads();
    for (int mm = 0; mm < 256; ++mm) {
      const float dx = sx[mm] - qx, dy = sy[mm] - qy, dz = sz[mm] - qz;
      const float d = dx * dx + dy * dy + dz * dz;
      if (d < wd) {
#pragma unroll
        for (int j = 0; j < kK; ++j)
          if (j == wpos) { bd[j] = d; bi[j] = m0 + mm; }
        wd = -1.0f;
#pragma unroll
        for (int j = 0; j < kK; ++j)
          if (bd[j] > wd) { wd = bd[j]; wpos = j; }
      }
    }
  }
#pragma unroll
  for (int j = 0; j < kK; ++j) outidx[(b * kN + n) * kK + j] = bi[j];
}

// ---------------------------------------------------------------------------
// Pack W[O,Cin] f32 -> bf16 in the per-lane WMMA A-fragment layout:
//   packed[((rt*KC + kc)*32 + lane)*16 + j]
// A k order used here (contiguous, as in the B loads): lane<16 holds row rt*16+lane, j=0..15 -> K 0..15 ;
// lane>=16 holds same row, j=0..15 -> K 16..31.  Zero-pad Cin up to KC*32.
// ---------------------------------------------------------------------------
__global__ void pack_w_kernel(const float* __restrict__ W, __bf16* __restrict__ out,
                              int O, int Cin, int KC) {
  const int tid = blockIdx.x * 256 + threadIdx.x;
  const int total = (O / 16) * KC * 32 * 16;
  if (tid >= total) return;
  const int j    = tid & 15;
  const int lane = (tid >> 4) & 31;
  const int kc   = (tid >> 9) % KC;
  const int rt   = tid / (KC << 9);
  const int o    = rt * 16 + (lane & 15);
  int h;
  if (lane < 16) h = j;
  else           h = j + 16;
  const int c = kc * 32 + h;
  const float v = (c < Cin) ? W[o * Cin + c] : 0.0f;
  out[tid] = (__bf16)v;
}

__global__ void zero_kernel(float* __restrict__ p, int n) {
  const int i = blockIdx.x * 256 + threadIdx.x;
  if (i < n) p[i] = 0.0f;
}

// ---------------------------------------------------------------------------
// Device helpers for the fused flow-embedding kernel.
// ---------------------------------------------------------------------------
template <int O, int KC, int CPAD>
__device__ __forceinline__ void gemm_tile(const __bf16* __restrict__ wp,
                                          const __bf16* xs, int colbase, int hi,
                                          int lane, v8f* acc) {
#pragma unroll
  for (int rt = 0; rt < O / 16; ++rt)
    acc[rt] = (v8f){0.f, 0.f, 0.f, 0.f, 0.f, 0.f, 0.f, 0.f};
#pragma unroll
  for (int kc = 0; kc < KC; ++kc) {
    // B fragment (32x16 bf16): lane<16 -> col=lane, K 0..15 ; lane>=16 -> K 16..31
    const v16bf bfr = *(const v16bf*)(xs + colbase * CPAD + kc * 32 + hi);
#pragma unroll
    for (int rt = 0; rt < O / 16; ++rt) {
      const v16bf afr = *(const v16bf*)(wp + ((rt * KC + kc) * 32 + lane) * 16);
      acc[rt] = __builtin_amdgcn_wmma_f32_16x16x32_bf16(
          false, afr, false, bfr, (short)0, acc[rt], false, false);
    }
  }
}

template <int O>
__device__ __forceinline__ void accum_stats(const v8f* acc, int lane, int osel,
                                            float* __restrict__ st) {
#pragma unroll
  for (int rt = 0; rt < O / 16; ++rt) {
#pragma unroll
    for (int v = 0; v < 8; ++v) {
      const float y = acc[rt][v];
      float s = y, q = y * y;
#pragma unroll
      for (int d = 1; d < 16; d <<= 1) {   // reduce within each 16-lane half
        s += __shfl_xor(s, d, 32);
        q += __shfl_xor(q, d, 32);
      }
      if ((lane & 15) == 0) {
        const int o = rt * 16 + v + osel;
        atomicAdd(&st[o], s);
        atomicAdd(&st[128 + o], q);
      }
    }
  }
}

template <int O, int CPAD>
__device__ __forceinline__ void bn_act_store(const v8f* acc, int lane, int colbase,
                                             int osel, const float* bA,
                                             const float* bB, __bf16* xs) {
#pragma unroll
  for (int rt = 0; rt < O / 16; ++rt) {
#pragma unroll
    for (int v = 0; v < 8; ++v) {
      const int o = rt * 16 + v + osel;
      float a = bA[o] * acc[rt][v] + bB[o];
      a = a > 0.0f ? a : 0.01f * a;
      xs[colbase * CPAD + o] = (__bf16)a;
    }
  }
}

__device__ __forceinline__ void bn_prep(int t, int O, const float* st, const float* g,
                                        const float* bb, float* A, float* Bv) {
  if (t < O) {
    const float mn = st[t] / kCnt;
    const float vr = st[128 + t] / kCnt - mn * mn;
    const float a  = g[t] * rsqrtf(vr + 1e-5f);
    A[t]  = a;
    Bv[t] = bb[t] - mn * a;
  }
}

// ---------------------------------------------------------------------------
// Fused flow-embedding pass.
//   DEPTH=1: gather + GEMM1            -> accumulate stats for BN1
//   DEPTH=2: ... + BN1 + GEMM2         -> stats for BN2
//   DEPTH=3: ... + BN2 + GEMM3         -> stats for BN3
//   DEPTH=4: ... + BN3 + lrelu + maxpool over K -> out [B,O3,N]
// Block = COLS*2 threads; COLS columns = COLS/32 query points x 32 neighbors.
// Each wave owns one 16-column tile -> all LDS reuse is wave-local.
// ---------------------------------------------------------------------------
template <int C, int O1, int O2, int O3, int COLS, int DEPTH>
__global__ __launch_bounds__(COLS * 2) void fe_kernel(
    const float* __restrict__ pos1, const float* __restrict__ pos2,
    const float* __restrict__ feat1, const float* __restrict__ feat2,
    const int* __restrict__ idx,
    const __bf16* __restrict__ wp1, const __bf16* __restrict__ wp2,
    const __bf16* __restrict__ wp3,
    const float* __restrict__ g1, const float* __restrict__ bb1,
    const float* __restrict__ g2, const float* __restrict__ bb2,
    const float* __restrict__ g3, const float* __restrict__ bb3,
    float* __restrict__ st1, float* __restrict__ st2, float* __restrict__ st3,
    float* __restrict__ out) {
  constexpr int CIN  = 2 * C + 3;
  constexpr int KC1  = (CIN + 31) / 32;
  constexpr int CPAD = KC1 * 32;
  constexpr int KC2  = O1 / 32;
  constexpr int KC3  = O2 / 32;
  constexpr int PTS  = COLS / 32;
  constexpr int THREADS = COLS * 2;

  __shared__ __align__(32) __bf16 xs[COLS * CPAD];
  __shared__ float bnA[3][128];
  __shared__ float bnB[3][128];
  __shared__ float pm[PTS][O3][2];

  const int t  = threadIdx.x;
  const int b  = blockIdx.x / (kN / PTS);
  const int n0 = (blockIdx.x % (kN / PTS)) * PTS;

  if constexpr (DEPTH >= 2) bn_prep(t, O1, st1, g1, bb1, bnA[0], bnB[0]);
  if constexpr (DEPTH >= 3) bn_prep(t, O2, st2, g2, bb2, bnA[1], bnB[1]);
  if constexpr (DEPTH >= 4) bn_prep(t, O3, st3, g3, bb3, bnA[2], bnB[2]);

  // ---- gather: build x[col][c] = concat(pos_diff, feat2[m], feat1[n]) ----
  {
    const int col  = t >> 1;
    const int half = t & 1;
    const int p = col >> 5, k = col & 31;
    const int n = n0 + p;
    const int m = idx[(b * kN + n) * kK + k];
    for (int c = half; c < CPAD; c += 2) {
      float v = 0.0f;
      if (c < 3)           v = pos2[(b * 3 + c) * kN + m] - pos1[(b * 3 + c) * kN + n];
      else if (c < 3 + C)  v = feat2[(b * C + (c - 3)) * kN + m];
      else if (c < CIN)    v = feat1[(b * C + (c - 3 - C)) * kN + n];
      xs[col * CPAD + c] = (__bf16)v;
    }
  }
  __syncthreads();

  const int wave = t >> 5, lane = t & 31;
  const int colbase = wave * 16 + (lane & 15);
  const int hi   = lane & 16;          // +16 halves for lanes 16..31 (B frag)
  const int osel = (lane >> 4) << 3;   // D frag: lane>=16 -> rows M=8..15

  // ---- layer 1 ----
  {
    v8f a1[O1 / 16];
    gemm_tile<O1, KC1, CPAD>(wp1, xs, colbase, hi, lane, a1);
    if constexpr (DEPTH == 1) { accum_stats<O1>(a1, lane, osel, st1); return; }
    bn_act_store<O1, CPAD>(a1, lane, colbase, osel, bnA[0], bnB[0], xs);
  }
  // ---- layer 2 ----
  {
    v8f a2[O2 / 16];
    gemm_tile<O2, KC2, CPAD>(wp2, xs, colbase, hi, lane, a2);
    if constexpr (DEPTH == 2) { accum_stats<O2>(a2, lane, osel, st2); return; }
    if constexpr (DEPTH >= 3)
      bn_act_store<O2, CPAD>(a2, lane, colbase, osel, bnA[1], bnB[1], xs);
  }
  // ---- layer 3 ----
  if constexpr (DEPTH >= 3) {
    v8f a3[O3 / 16];
    gemm_tile<O3, KC3, CPAD>(wp3, xs, colbase, hi, lane, a3);
    if constexpr (DEPTH == 3) { accum_stats<O3>(a3, lane, osel, st3); return; }
    // DEPTH == 4: BN3 + leaky-relu + max over neighbors
#pragma unroll
    for (int rt = 0; rt < O3 / 16; ++rt) {
#pragma unroll
      for (int v = 0; v < 8; ++v) {
        const int o = rt * 16 + v + osel;
        float a = bnA[2][o] * a3[rt][v] + bnB[2][o];
        a = a > 0.0f ? a : 0.01f * a;
        float mx = a;
#pragma unroll
        for (int d = 1; d < 16; d <<= 1) mx = fmaxf(mx, __shfl_xor(mx, d, 32));
        if ((lane & 15) == 0) pm[wave >> 1][o][wave & 1] = mx;
      }
    }
    __syncthreads();
    for (int i = t; i < PTS * O3; i += THREADS) {
      const int p = i / O3, o = i % O3;
      out[(b * O3 + o) * kN + n0 + p] = fmaxf(pm[p][o][0], pm[p][o][1]);
    }
  }
}

// ---------------------------------------------------------------------------
// Host launch
// ---------------------------------------------------------------------------
extern "C" void kernel_launch(void* const* d_in, const int* in_sizes, int n_in,
                              void* d_out, int out_size, void* d_ws, size_t ws_size,
                              hipStream_t stream) {
  const float* f0 = (const float*)d_in[0];
  const float* f1 = (const float*)d_in[1];
  const float* f2 = (const float*)d_in[2];
  const float* p0 = (const float*)d_in[3];
  const float* p1 = (const float*)d_in[4];
  const float* p2 = (const float*)d_in[5];
  const float* w00 = (const float*)d_in[7],  *g00 = (const float*)d_in[8],  *b00 = (const float*)d_in[9];
  const float* w01 = (const float*)d_in[10], *g01 = (const float*)d_in[11], *b01 = (const float*)d_in[12];
  const float* w02 = (const float*)d_in[13], *g02 = (const float*)d_in[14], *b02 = (const float*)d_in[15];
  const float* w10 = (const float*)d_in[16], *g10 = (const float*)d_in[17], *b10 = (const float*)d_in[18];
  const float* w11 = (const float*)d_in[19], *g11 = (const float*)d_in[20], *b11 = (const float*)d_in[21];
  const float* w12 = (const float*)d_in[22], *g12 = (const float*)d_in[23], *b12 = (const float*)d_in[24];

  char* ws = (char*)d_ws;
  // workspace layout
  int* idx01 = (int*)ws;                                   // 2 MB
  int* idx12 = idx01 + kNB * kN * kK;                      // 2 MB
  float* h01 = (float*)(ws + 4 * 1024 * 1024);             // 8 MB  [B,128,N]
  float* h12 = h01 + kNB * 128 * kN;                       // 8 MB
  __bf16* wpb = (__bf16*)(ws + 20971520);                  // packed weights
  const int S00 = 4 * 5 * 512, S01 = 4 * 2 * 512, S02 = 8 * 2 * 512;
  const int S10 = 8 * 9 * 512, S11 = 8 * 4 * 512, S12 = 8 * 4 * 512;
  __bf16* wp00 = wpb;
  __bf16* wp01 = wp00 + S00;
  __bf16* wp02 = wp01 + S01;
  __bf16* wp10 = wp02 + S02;
  __bf16* wp11 = wp10 + S10;
  __bf16* wp12 = wp11 + S11;
  float* stats = (float*)(ws + 20971520 + 2 * (S00 + S01 + S02 + S10 + S11 + S12));
  auto st = [&](int fe, int l) { return stats + (fe * 3 + l) * 256; };

  // 1) pack weights into WMMA A-fragment layout (bf16)
  pack_w_kernel<<<(S00 + 255) / 256, 256, 0, stream>>>(w00, wp00, 64, 131, 5);
  pack_w_kernel<<<(S01 + 255) / 256, 256, 0, stream>>>(w01, wp01, 64, 64, 2);
  pack_w_kernel<<<(S02 + 255) / 256, 256, 0, stream>>>(w02, wp02, 128, 64, 2);
  pack_w_kernel<<<(S10 + 255) / 256, 256, 0, stream>>>(w10, wp10, 128, 259, 9);
  pack_w_kernel<<<(S11 + 255) / 256, 256, 0, stream>>>(w11, wp11, 128, 128, 4);
  pack_w_kernel<<<(S12 + 255) / 256, 256, 0, stream>>>(w12, wp12, 128, 128, 4);

  // 2) zero BN stats accumulators (deterministic per call)
  zero_kernel<<<(9 * 256 + 255) / 256, 256, 0, stream>>>(stats, 9 * 256);

  // 3) KNN (idx01 shared by FE1 and FE3)
  knn_kernel<<<64, 256, 0, stream>>>(p0, p1, idx01);
  knn_kernel<<<64, 256, 0, stream>>>(p1, p2, idx12);

  // 4) flow embeddings
#define FE_L0(D, PA, PB, FA, FB, IDX, FEI, OUT)                                   \
  fe_kernel<64, 64, 64, 128, 128, D><<<kNB * kN / 4, 256, 0, stream>>>(           \
      PA, PB, FA, FB, IDX, wp00, wp01, wp02, g00, b00, g01, b01, g02, b02,        \
      st(FEI, 0), st(FEI, 1), st(FEI, 2), OUT)
#define FE_L1(D, PA, PB, FA, FB, IDX, FEI, OUT)                                   \
  fe_kernel<128, 128, 128, 128, 64, D><<<kNB * kN / 2, 128, 0, stream>>>(         \
      PA, PB, FA, FB, IDX, wp10, wp11, wp12, g10, b10, g11, b11, g12, b12,        \
      st(FEI, 0), st(FEI, 1), st(FEI, 2), OUT)

  // FE1: (p0, p1, f0, f1) -> h01
  FE_L0(1, p0, p1, f0, f1, idx01, 0, h01);
  FE_L0(2, p0, p1, f0, f1, idx01, 0, h01);
  FE_L0(3, p0, p1, f0, f1, idx01, 0, h01);
  FE_L0(4, p0, p1, f0, f1, idx01, 0, h01);
  // FE2: (p1, p2, f1, f2) -> h12
  FE_L0(1, p1, p2, f1, f2, idx12, 1, h12);
  FE_L0(2, p1, p2, f1, f2, idx12, 1, h12);
  FE_L0(3, p1, p2, f1, f2, idx12, 1, h12);
  FE_L0(4, p1, p2, f1, f2, idx12, 1, h12);
  // FE3: (p0, p1, h01, h12) -> out
  float* outf = (float*)d_out;
  FE_L1(1, p0, p1, h01, h12, idx01, 2, outf);
  FE_L1(2, p0, p1, h01, h12, idx01, 2, outf);
  FE_L1(3, p0, p1, h01, h12, idx01, 2, outf);
  FE_L1(4, p0, p1, h01, h12, idx01, 2, outf);

#undef FE_L0
#undef FE_L1
  (void)in_sizes; (void)n_in; (void)out_size; (void)ws_size;
}
